// GPRGNN_27255862460747
// MI455X (gfx1250) — matrix operand order repaired and checked by reading, not compiled, not run on hardware
//
#include <hip/hip_runtime.h>
#include <math.h>

typedef __attribute__((ext_vector_type(16))) _Float16 v16h;
typedef __attribute__((ext_vector_type(8)))  _Float16 v8h;
typedef __attribute__((ext_vector_type(8)))  float    v8f;

#define NN      50000
#define IN_DIM  256
#define HIDDEN  256
#define OUT_DIM 48
#define K_HOPS  10

// -------------------- degree / normalization --------------------
__global__ void fill_ones_k(float* __restrict__ p, int n) {
    int i = blockIdx.x * blockDim.x + threadIdx.x;
    if (i < n) p[i] = 1.0f;   // self-loop contributes 1 to every degree
}

__global__ void deg_count_k(const long long* __restrict__ ei, int E, float* __restrict__ deg) {
    int e = blockIdx.x * blockDim.x + threadIdx.x;
    if (e < E) atomicAdd(&deg[(int)ei[(size_t)E + e]], 1.0f);
}

__global__ void deg_to_dinv_k(float* __restrict__ d, int n) {
    int i = blockIdx.x * blockDim.x + threadIdx.x;
    if (i < n) d[i] = rsqrtf(fmaxf(d[i], 1e-12f));
}

// -------------------- MLP layer 1: Hh = relu(X@W1+b1), f16 out --------------------
// grid.x = NN/16, block = 512 (16 waves, one 16-wide N tile each)
__global__ void mlp1_wmma_k(const float* __restrict__ X, const float* __restrict__ W1,
                            const float* __restrict__ b1, _Float16* __restrict__ Hh) {
    const int wave = threadIdx.x >> 5;
    const int lane = threadIdx.x & 31;
    const int m    = lane & 15;
    const int hi   = lane >> 4;
    const int rb   = blockIdx.x * 16;
    const int n0   = wave * 16;
    const float* arow = X + (size_t)(rb + m) * IN_DIM;
    const int kb_lane = m + hi * 16;           // unused; this lane supplies B column n0+m (see B tile below)

    v8f c = {};
#pragma unroll
    for (int kt = 0; kt < IN_DIM / 32; ++kt) {
        const int kBase = kt * 32;
        // A tile: K = hi*8..hi*8+7 and 16+hi*8..16+hi*8+7 (ISA 16-bit A layout)
        const float4 a0 = *(const float4*)(arow + kBase + hi * 8);
        const float4 a1 = *(const float4*)(arow + kBase + hi * 8 + 4);
        const float4 a2 = *(const float4*)(arow + kBase + 16 + hi * 8);
        const float4 a3 = *(const float4*)(arow + kBase + 16 + hi * 8 + 4);
        v16h A;
        A[0]=(_Float16)a0.x; A[1]=(_Float16)a0.y; A[2]=(_Float16)a0.z; A[3]=(_Float16)a0.w;
        A[4]=(_Float16)a1.x; A[5]=(_Float16)a1.y; A[6]=(_Float16)a1.z; A[7]=(_Float16)a1.w;
        A[8]=(_Float16)a2.x; A[9]=(_Float16)a2.y; A[10]=(_Float16)a2.z; A[11]=(_Float16)a2.w;
        A[12]=(_Float16)a3.x; A[13]=(_Float16)a3.y; A[14]=(_Float16)a3.z; A[15]=(_Float16)a3.w;
        // B tile: lane holds column N=n0+m, K = hi*8..hi*8+7 and 16+hi*8..16+hi*8+7
        const float* bcol = W1 + (size_t)(kBase + hi * 8) * HIDDEN + n0 + m;
        v16h B;
#pragma unroll
        for (int j = 0; j < 8; ++j) { B[j] = (_Float16)bcol[(size_t)j * HIDDEN]; B[8 + j] = (_Float16)bcol[(size_t)(16 + j) * HIDDEN]; }

        c = __builtin_amdgcn_wmma_f32_16x16x32_f16(false, A, false, B,
                                                   (short)0, c, false, false);
    }
    const float bias = b1[n0 + m];             // C/D: N = lane&15
#pragma unroll
    for (int r = 0; r < 8; ++r) {
        const int mm = hi * 8 + r;             // C/D: M = r (+8 for upper half-wave)
        float v = c[r] + bias;
        v = v > 0.0f ? v : 0.0f;
        Hh[(size_t)(rb + mm) * HIDDEN + n0 + m] = (_Float16)v;
    }
}

// -------------------- MLP layer 2: H2 = Hh@W2+b2, f32 out --------------------
// grid.x = NN/16, block = 96 (3 waves: N tiles 0,16,32)
__global__ void mlp2_wmma_k(const _Float16* __restrict__ Hh, const float* __restrict__ W2,
                            const float* __restrict__ b2, float* __restrict__ H2) {
    const int wave = threadIdx.x >> 5;
    const int lane = threadIdx.x & 31;
    const int m    = lane & 15;
    const int hi   = lane >> 4;
    const int rb   = blockIdx.x * 16;
    const int n0   = wave * 16;
    const _Float16* arow = Hh + (size_t)(rb + m) * HIDDEN;
    const int kb_lane = m + hi * 16;

    v8f c = {};
#pragma unroll
    for (int kt = 0; kt < HIDDEN / 32; ++kt) {
        const int kBase = kt * 32;
        const v8h a0 = *(const v8h*)(arow + kBase + hi * 8);
        const v8h a1 = *(const v8h*)(arow + kBase + 16 + hi * 8);
        v16h A;
#pragma unroll
        for (int j = 0; j < 8; ++j) { A[j] = a0[j]; A[8 + j] = a1[j]; }

        const float* bcol = W2 + (size_t)(kBase + hi * 8) * OUT_DIM + n0 + m;
        v16h B;
#pragma unroll
        for (int j = 0; j < 8; ++j) { B[j] = (_Float16)bcol[(size_t)j * OUT_DIM]; B[8 + j] = (_Float16)bcol[(size_t)(16 + j) * OUT_DIM]; }

        c = __builtin_amdgcn_wmma_f32_16x16x32_f16(false, A, false, B,
                                                   (short)0, c, false, false);
    }
    const float bias = b2[n0 + m];
#pragma unroll
    for (int r = 0; r < 8; ++r) {
        const int mm = hi * 8 + r;
        H2[(size_t)(rb + mm) * OUT_DIM + n0 + m] = c[r] + bias;
    }
}

// -------------------- propagation --------------------
__global__ void scale_init_k(float* __restrict__ hidden, const float* __restrict__ h,
                             const float* __restrict__ temp, int n) {
    int i = blockIdx.x * blockDim.x + threadIdx.x;
    if (i < n) hidden[i] = temp[0] * h[i];
}

// next = dinv[node]^2 * prev   (self-loop contribution; also initializes next, so no zeroing is needed)
__global__ void selfloop_init_k(const float* __restrict__ prev, const float* __restrict__ dinv,
                                float* __restrict__ next, int n) {
    int i = blockIdx.x * blockDim.x + threadIdx.x;
    if (i < n) {
        int node = i / OUT_DIM;
        float di = dinv[node];
        next[i] = di * di * prev[i];
    }
}

// one thread per (edge, dim)
__global__ void edge_prop_k(const long long* __restrict__ ei, int E,
                            const float* __restrict__ dinv,
                            const float* __restrict__ prev, float* __restrict__ next) {
    long long idx = (long long)blockIdx.x * blockDim.x + threadIdx.x;
    long long total = (long long)E * OUT_DIM;
    if (idx < total) {
        int e = (int)(idx / OUT_DIM);
        int d = (int)(idx % OUT_DIM);
        int s = (int)ei[e];
        int t = (int)ei[(size_t)E + e];
        float w = dinv[s] * dinv[t];
        atomicAdd(&next[(size_t)t * OUT_DIM + d], w * prev[(size_t)s * OUT_DIM + d]);
    }
}

__global__ void accum_k(float* __restrict__ hidden, const float* __restrict__ hk,
                        const float* __restrict__ temp, int k, int n) {
    int i = blockIdx.x * blockDim.x + threadIdx.x;
    if (i < n) hidden[i] += temp[k] * hk[i];
}

// -------------------- log_softmax over 48 dims, in place --------------------
__global__ void log_softmax_k(float* __restrict__ out, int n) {
    int node = blockIdx.x * blockDim.x + threadIdx.x;
    if (node < n) {
        float* row = out + (size_t)node * OUT_DIM;
        float mx = -INFINITY;
#pragma unroll
        for (int d = 0; d < OUT_DIM; ++d) mx = fmaxf(mx, row[d]);
        float s = 0.0f;
#pragma unroll
        for (int d = 0; d < OUT_DIM; ++d) s += expf(row[d] - mx);
        const float ls = mx + logf(s);
#pragma unroll
        for (int d = 0; d < OUT_DIM; ++d) row[d] = row[d] - ls;
    }
}

extern "C" void kernel_launch(void* const* d_in, const int* in_sizes, int n_in,
                              void* d_out, int out_size, void* d_ws, size_t ws_size,
                              hipStream_t stream) {
    const float*     X    = (const float*)d_in[0];
    const long long* EI   = (const long long*)d_in[1];
    const float*     W1   = (const float*)d_in[2];
    const float*     b1   = (const float*)d_in[3];
    const float*     W2   = (const float*)d_in[4];
    const float*     b2   = (const float*)d_in[5];
    const float*     temp = (const float*)d_in[6];
    const int E = in_sizes[1] / 2;

    // workspace carve-up (~45 MB)
    char* ws = (char*)d_ws;
    _Float16* Hh  = (_Float16*)ws; ws += (size_t)NN * HIDDEN * sizeof(_Float16);
    float*    hkA = (float*)ws;    ws += (size_t)NN * OUT_DIM * sizeof(float);
    float*    hkB = (float*)ws;    ws += (size_t)NN * OUT_DIM * sizeof(float);
    float*    dinv= (float*)ws;    ws += (size_t)NN * sizeof(float);
    float*    out = (float*)d_out;                 // doubles as "hidden" accumulator

    const int NH = NN * OUT_DIM;                   // 2,400,000

    // 1) degree -> dinv
    fill_ones_k<<<(NN + 255) / 256, 256, 0, stream>>>(dinv, NN);
    deg_count_k<<<(E + 255) / 256, 256, 0, stream>>>(EI, E, dinv);
    deg_to_dinv_k<<<(NN + 255) / 256, 256, 0, stream>>>(dinv, NN);

    // 2) MLP head on matrix cores
    mlp1_wmma_k<<<NN / 16, 512, 0, stream>>>(X, W1, b1, Hh);
    mlp2_wmma_k<<<NN / 16, 96, 0, stream>>>(Hh, W2, b2, hkA);   // hkA = h (hop 0)

    // 3) hidden = temp[0] * h ; then K hops of normalized propagation
    scale_init_k<<<(NH + 255) / 256, 256, 0, stream>>>(out, hkA, temp, NH);
    float* prev = hkA;
    float* next = hkB;
    const long long etotal = (long long)E * OUT_DIM;
    const int eblocks = (int)((etotal + 255) / 256);
    for (int k = 1; k <= K_HOPS; ++k) {
        selfloop_init_k<<<(NH + 255) / 256, 256, 0, stream>>>(prev, dinv, next, NH);
        edge_prop_k<<<eblocks, 256, 0, stream>>>(EI, E, dinv, prev, next);
        accum_k<<<(NH + 255) / 256, 256, 0, stream>>>(out, next, temp, k, NH);
        float* t = prev; prev = next; next = t;
    }

    // 4) log_softmax in place on d_out
    log_softmax_k<<<(NN + 255) / 256, 256, 0, stream>>>(out, NN);
}
